// MLRelResidualClassifier_81879256531384
// MI455X (gfx1250) — compile-verified, matrix operand order checked by reading, not run on hardware
//
#include <hip/hip_runtime.h>
#include <hip/hip_bf16.h>

// Problem constants (match reference)
#define B_   16
#define L_   256
#define H_   768
#define N_   1024
#define E_   4096
#define R_   40
#define C_   36
#define NPB  (N_ / B_)        // nodes per batch (node_batch is repeat(arange(B), N/B))
#define KT_  (H_ / 32)        // 24 K-steps of 32 (bf16 WMMA K)
#define NT_  (H_ / 16)        // 48 column tiles
#define CT_  4                // column tiles per wave (64-wide output strip)
#define NG_  (NT_ / CT_)      // 12 column groups
#define RT_  2                // row tiles per wave (32 rows)
#define MG_  (N_ / (RT_ * 16))// 32 row groups for root GEMM
#define MAXTILES (E_ / 32 + R_)

typedef __attribute__((ext_vector_type(16))) __bf16 v16bf;
typedef __attribute__((ext_vector_type(8)))  float  v8f;

// A fragment (16x32 bf16, ISA §7.12.2 layout) for one lane from a row-major
// f32 row: lane<16 holds K {k0..k0+7, k0+16..k0+23}, lane>=16 holds
// K {k0+8..k0+15, k0+24..k0+31}. Native casts -> v_cvt_pk_bf16_f32.
__device__ __forceinline__ v16bf load_a_frag(const float* __restrict__ xrow,
                                             int k0, int half) {
  const float4* pa0 = (const float4*)(xrow + k0 + half);
  const float4* pa1 = (const float4*)(xrow + k0 + 16 + half);
  float4 f0 = pa0[0], f1 = pa0[1], f2 = pa1[0], f3 = pa1[1];
  v16bf a;
  a[0]  = (__bf16)f0.x; a[1]  = (__bf16)f0.y; a[2]  = (__bf16)f0.z; a[3]  = (__bf16)f0.w;
  a[4]  = (__bf16)f1.x; a[5]  = (__bf16)f1.y; a[6]  = (__bf16)f1.z; a[7]  = (__bf16)f1.w;
  a[8]  = (__bf16)f2.x; a[9]  = (__bf16)f2.y; a[10] = (__bf16)f2.z; a[11] = (__bf16)f2.w;
  a[12] = (__bf16)f3.x; a[13] = (__bf16)f3.y; a[14] = (__bf16)f3.z; a[15] = (__bf16)f3.w;
  return a;
}

// B fragment (32x16 bf16, column nCol) from row-major f32 W[k][n].
__device__ __forceinline__ v16bf load_b_frag(const float* __restrict__ W,
                                             int k0, int half, int nCol) {
  v16bf b;
  #pragma unroll
  for (int j = 0; j < 8; ++j) {
    b[j]     = (__bf16)W[(size_t)(k0 + half + j) * H_ + nCol];
    b[j + 8] = (__bf16)W[(size_t)(k0 + 16 + half + j) * H_ + nCol];
  }
  return b;
}

#define WMMA_BF16(A, Bf, Cacc) \
  __builtin_amdgcn_wmma_f32_16x16x32_bf16(false, (A), false, (Bf), (short)0, (Cacc), false, false)

// ---------------- entity extraction: e1_h/e2_h [B,H] ----------------
__global__ void k_entity(const float* __restrict__ seq, const float* __restrict__ m1,
                         const float* __restrict__ m2, float* __restrict__ e1h,
                         float* __restrict__ e2h) {
  int idx = blockIdx.x * blockDim.x + threadIdx.x;
  if (idx >= B_ * H_) return;
  int b = idx / H_, h = idx - b * H_;
  const float* s = seq + (size_t)b * L_ * H_ + h;
  float a1 = 0.f, a2 = 0.f;
  for (int l = 0; l < L_; ++l) {
    float v = s[(size_t)l * H_];
    a1 += m1[b * L_ + l] * v;
    a2 += m2[b * L_ + l] * v;
  }
  e1h[idx] = a1; e2h[idx] = a2;
}

// ---------------- node embeddings: max over L of seq[nb] + dep_x ----------------
__global__ void k_graph_embs(const float* __restrict__ seq, const float* __restrict__ dep,
                             const int* __restrict__ node_batch, float* __restrict__ ge) {
  __shared__ float sdep[L_];
  int n = blockIdx.y;
  int h = blockIdx.x * blockDim.x + threadIdx.x;
  for (int l = threadIdx.x; l < L_; l += blockDim.x) sdep[l] = dep[n * L_ + l];
  __syncthreads();
  int nb = node_batch[n];
  const float* s = seq + (size_t)nb * L_ * H_ + h;
  float m = -3.4e38f;
  for (int l = 0; l < L_; ++l) m = fmaxf(m, s[(size_t)l * H_] + sdep[l]);
  ge[(size_t)n * H_ + h] = m;
}

// ---------------- per-(dst,rel) degree counts + per-rel totals ----------------
__global__ void k_count(const int* __restrict__ ei, const int* __restrict__ et,
                        int* __restrict__ cnt, int* __restrict__ relcnt) {
  int e = blockIdx.x * blockDim.x + threadIdx.x;
  if (e >= E_) return;
  int d = ei[E_ + e], r = et[e];
  atomicAdd(&cnt[d * R_ + r], 1);
  atomicAdd(&relcnt[r], 1);
}

// ---------------- serial scan (40 entries) + 32-edge tile worklist ----------------
__global__ void k_scan(const int* __restrict__ relcnt, int* __restrict__ offs,
                       int* __restrict__ tiles, int* __restrict__ ntl,
                       int* __restrict__ cursor) {
  if (threadIdx.x != 0 || blockIdx.x != 0) return;
  int off = 0, nt = 0;
  for (int r = 0; r < R_; ++r) {
    offs[r] = off; cursor[r] = 0;
    int c = relcnt[r];
    int t = (c + 31) >> 5;
    for (int i = 0; i < t; ++i) tiles[nt++] = (r << 16) | i;
    off += c;
  }
  offs[R_] = off;
  *ntl = nt;
}

// ---------------- bucket edges by relation; per-edge norm ----------------
__global__ void k_scatter(const int* __restrict__ ei, const int* __restrict__ et,
                          const int* __restrict__ cnt, const int* __restrict__ offs,
                          int* __restrict__ cursor, int* __restrict__ perm,
                          float* __restrict__ norm) {
  int e = blockIdx.x * blockDim.x + threadIdx.x;
  if (e >= E_) return;
  int r = et[e], d = ei[E_ + e];
  int pos = offs[r] + atomicAdd(&cursor[r], 1);
  perm[pos] = e;
  int c = cnt[d * R_ + r];
  norm[e] = 1.0f / (float)(c > 1 ? c : 1);
}

// ---------------- edge-message GEMM: msg = x[src] @ W_rel[r], scatter-add ----------------
// One wave per 32-edge x 64-col tile: 8 WMMAs per K-step, B panel shared by 2 row tiles.
__global__ __launch_bounds__(32)
void k_edge_gemm(const float* __restrict__ x, const float* __restrict__ Wrel,
                 const int* __restrict__ ei, const int* __restrict__ perm,
                 const int* __restrict__ offs, const int* __restrict__ tiles,
                 const int* __restrict__ ntl, const float* __restrict__ norm,
                 float* __restrict__ agg) {
  int tileId = blockIdx.y;
  if (tileId >= *ntl) return;
  int td = tiles[tileId];
  int r = td >> 16, t = td & 0xFFFF;
  int base = offs[r] + t * 32;
  int mval = offs[r + 1] - offs[r] - t * 32;
  if (mval > 32) mval = 32;

  __shared__ int   sSrc[32];
  __shared__ int   sDst[32];
  __shared__ float sNrm[32];
  int lane = threadIdx.x;
  if (lane < mval) {
    int e = perm[base + lane];
    sSrc[lane] = ei[e];
    sDst[lane] = ei[E_ + e];
    sNrm[lane] = norm[e];
  } else { sSrc[lane] = 0; sDst[lane] = -1; sNrm[lane] = 0.f; }
  __syncthreads();

  const float* W = Wrel + (size_t)r * H_ * H_;
  int m     = lane & 15;
  int half  = (lane >> 4) ? 8 : 0;            // K sub-block per lane-half (ISA A layout)
  int nBase = blockIdx.x * (CT_ * 16) + m;    // this lane's first B column
  const float* xrow0 = x + (size_t)sSrc[m] * H_;       // row tile 0 (edges 0..15)
  const float* xrow1 = x + (size_t)sSrc[m + 16] * H_;  // row tile 1 (edges 16..31)

  v8f p00 = {}, p01 = {}, p02 = {}, p03 = {};
  v8f p10 = {}, p11 = {}, p12 = {}, p13 = {};
  for (int kt = 0; kt < KT_; ++kt) {
    int k0 = kt * 32;
    v16bf a0 = load_a_frag(xrow0, k0, half);
    v16bf a1 = load_a_frag(xrow1, k0, half);
    v16bf b0 = load_b_frag(W, k0, half, nBase);
    v16bf b1 = load_b_frag(W, k0, half, nBase + 16);
    v16bf b2 = load_b_frag(W, k0, half, nBase + 32);
    v16bf b3 = load_b_frag(W, k0, half, nBase + 48);
    p00 = WMMA_BF16(a0, b0, p00); p10 = WMMA_BF16(a1, b0, p10);
    p01 = WMMA_BF16(a0, b1, p01); p11 = WMMA_BF16(a1, b1, p11);
    p02 = WMMA_BF16(a0, b2, p02); p12 = WMMA_BF16(a1, b2, p12);
    p03 = WMMA_BF16(a0, b3, p03); p13 = WMMA_BF16(a1, b3, p13);
  }

  // C layout: vgpr v <-> row v + 8*(lane>=16); col = lane&15 within each tile
  int rowoff = (lane >> 4) ? 8 : 0;
  #pragma unroll
  for (int v = 0; v < 8; ++v) {
    int d0 = sDst[v + rowoff];
    if (d0 >= 0) {
      float* row = agg + (size_t)d0 * H_ + nBase;
      float nr = sNrm[v + rowoff];
      atomicAdd(row +  0, p00[v] * nr);
      atomicAdd(row + 16, p01[v] * nr);
      atomicAdd(row + 32, p02[v] * nr);
      atomicAdd(row + 48, p03[v] * nr);
    }
    int d1 = sDst[16 + v + rowoff];
    if (d1 >= 0) {
      float* row = agg + (size_t)d1 * H_ + nBase;
      float nr = sNrm[16 + v + rowoff];
      atomicAdd(row +  0, p10[v] * nr);
      atomicAdd(row + 16, p11[v] * nr);
      atomicAdd(row + 32, p12[v] * nr);
      atomicAdd(row + 48, p13[v] * nr);
    }
  }
}

// ---------------- root GEMM + fused epilogue: out = relu(agg + x@Wroot + b) ----------------
// One wave per 32-row x 64-col tile: 8 WMMAs per K-step.
__global__ __launch_bounds__(32)
void k_root_gemm(const float* __restrict__ x, const float* __restrict__ Wroot,
                 const float* __restrict__ bias, const float* __restrict__ agg,
                 float* __restrict__ out) {
  int lane  = threadIdx.x;
  int m     = lane & 15;
  int half  = (lane >> 4) ? 8 : 0;
  int row0  = blockIdx.y * (RT_ * 16);
  int nBase = blockIdx.x * (CT_ * 16) + m;
  const float* xrow0 = x + (size_t)(row0 + m) * H_;
  const float* xrow1 = x + (size_t)(row0 + 16 + m) * H_;

  v8f p00 = {}, p01 = {}, p02 = {}, p03 = {};
  v8f p10 = {}, p11 = {}, p12 = {}, p13 = {};
  for (int kt = 0; kt < KT_; ++kt) {
    int k0 = kt * 32;
    v16bf a0 = load_a_frag(xrow0, k0, half);
    v16bf a1 = load_a_frag(xrow1, k0, half);
    v16bf b0 = load_b_frag(Wroot, k0, half, nBase);
    v16bf b1 = load_b_frag(Wroot, k0, half, nBase + 16);
    v16bf b2 = load_b_frag(Wroot, k0, half, nBase + 32);
    v16bf b3 = load_b_frag(Wroot, k0, half, nBase + 48);
    p00 = WMMA_BF16(a0, b0, p00); p10 = WMMA_BF16(a1, b0, p10);
    p01 = WMMA_BF16(a0, b1, p01); p11 = WMMA_BF16(a1, b1, p11);
    p02 = WMMA_BF16(a0, b2, p02); p12 = WMMA_BF16(a1, b2, p12);
    p03 = WMMA_BF16(a0, b3, p03); p13 = WMMA_BF16(a1, b3, p13);
  }

  float bi0 = bias[nBase], bi1 = bias[nBase + 16], bi2 = bias[nBase + 32], bi3 = bias[nBase + 48];
  int rowoff = (lane >> 4) ? 8 : 0;
  #pragma unroll
  for (int v = 0; v < 8; ++v) {
    int gm0 = row0 + v + rowoff;
    const float* ar = agg + (size_t)gm0 * H_ + nBase;
    float* orow = out + (size_t)gm0 * H_ + nBase;
    orow[ 0] = fmaxf(p00[v] + ar[ 0] + bi0, 0.f);
    orow[16] = fmaxf(p01[v] + ar[16] + bi1, 0.f);
    orow[32] = fmaxf(p02[v] + ar[32] + bi2, 0.f);
    orow[48] = fmaxf(p03[v] + ar[48] + bi3, 0.f);
    int gm1 = gm0 + 16;
    const float* ar1 = agg + (size_t)gm1 * H_ + nBase;
    float* orow1 = out + (size_t)gm1 * H_ + nBase;
    orow1[ 0] = fmaxf(p10[v] + ar1[ 0] + bi0, 0.f);
    orow1[16] = fmaxf(p11[v] + ar1[16] + bi1, 0.f);
    orow1[32] = fmaxf(p12[v] + ar1[32] + bi2, 0.f);
    orow1[48] = fmaxf(p13[v] + ar1[48] + bi3, 0.f);
  }
}

// ---------------- masked node pooling into [B,H] ----------------
__global__ void k_dep_pool(const float* __restrict__ h, const float* __restrict__ n1,
                           const float* __restrict__ n2, float* __restrict__ d1,
                           float* __restrict__ d2) {
  int idx = blockIdx.x * blockDim.x + threadIdx.x;
  if (idx >= B_ * H_) return;
  int b = idx / H_, hh = idx - b * H_;
  float a1 = 0.f, a2 = 0.f;
  for (int i = 0; i < NPB; ++i) {
    int n = b * NPB + i;
    float v = h[(size_t)n * H_ + hh];
    a1 += n1[n] * v;
    a2 += n2[n] * v;
  }
  d1[idx] = a1; d2[idx] = a2;
}

// ---------------- rel_output = tanh(concat(ctx, e1h+e1d, e2h+e2d)) ----------------
__global__ void k_final(const float* __restrict__ ctx, const float* __restrict__ e1h,
                        const float* __restrict__ e1d, const float* __restrict__ e2h,
                        const float* __restrict__ e2d, float* __restrict__ out) {
  int idx = blockIdx.x * blockDim.x + threadIdx.x;
  if (idx >= B_ * 3 * H_) return;
  int b = idx / (3 * H_), j = idx - b * (3 * H_);
  float v;
  if (j < H_)            v = ctx[b * H_ + j];
  else if (j < 2 * H_) { int k = j - H_;     v = e1h[b * H_ + k] + e1d[b * H_ + k]; }
  else                 { int k = j - 2 * H_; v = e2h[b * H_ + k] + e2d[b * H_ + k]; }
  out[idx] = tanhf(v);
}

// ---------------- classifier logits [B,C] (tiny; VALU) ----------------
__global__ void k_logits(const float* __restrict__ ro, const float* __restrict__ Wc,
                         const float* __restrict__ bc, float* __restrict__ out) {
  int idx = blockIdx.x * blockDim.x + threadIdx.x;
  if (idx >= B_ * C_) return;
  int b = idx / C_, c = idx - b * C_;
  float a = bc[c];
  const float* row = ro + (size_t)b * 3 * H_;
  for (int k = 0; k < 3 * H_; ++k) a += row[k] * Wc[(size_t)k * C_ + c];
  out[idx] = a;
}

extern "C" void kernel_launch(void* const* d_in, const int* in_sizes, int n_in,
                              void* d_out, int out_size, void* d_ws, size_t ws_size,
                              hipStream_t stream) {
  const float* seq    = (const float*)d_in[0];
  const float* pooled = (const float*)d_in[1];
  const float* e1m    = (const float*)d_in[2];
  const float* e2m    = (const float*)d_in[3];
  const float* dep    = (const float*)d_in[4];
  const float* n1     = (const float*)d_in[5];
  const float* n2     = (const float*)d_in[6];
  const int*   ei     = (const int*)d_in[7];
  const int*   et     = (const int*)d_in[8];
  const int*   nb     = (const int*)d_in[9];
  const float* Wr1    = (const float*)d_in[10];
  const float* Wrel1  = (const float*)d_in[11];
  const float* b1     = (const float*)d_in[12];
  const float* Wr2    = (const float*)d_in[13];
  const float* Wrel2  = (const float*)d_in[14];
  const float* b2     = (const float*)d_in[15];
  const float* Wc     = (const float*)d_in[16];
  const float* bc     = (const float*)d_in[17];

  char* p = (char*)d_ws;
  auto alloc = [&](size_t bytes) -> void* {
    void* q = (void*)p;
    p += (bytes + 255) & ~(size_t)255;
    return q;
  };
  float* x0     = (float*)alloc((size_t)N_ * H_ * 4);   // node embs / layer-2 output
  float* h1     = (float*)alloc((size_t)N_ * H_ * 4);   // layer-1 output
  float* agg    = (float*)alloc((size_t)N_ * H_ * 4);   // message accumulator
  int*   cnt    = (int*)alloc((size_t)N_ * R_ * 4);
  int*   relcnt = (int*)alloc(R_ * 4);
  int*   offs   = (int*)alloc((R_ + 1) * 4);
  int*   cursor = (int*)alloc(R_ * 4);
  int*   perm   = (int*)alloc(E_ * 4);
  int*   tiles  = (int*)alloc(MAXTILES * 4);
  int*   ntl    = (int*)alloc(4);
  float* norm   = (float*)alloc(E_ * 4);
  float* e1h    = (float*)alloc((size_t)B_ * H_ * 4);
  float* e2h    = (float*)alloc((size_t)B_ * H_ * 4);
  float* e1d    = (float*)alloc((size_t)B_ * H_ * 4);
  float* e2d    = (float*)alloc((size_t)B_ * H_ * 4);

  float* rel_out = (float*)d_out;
  float* logits  = rel_out + (size_t)B_ * 3 * H_;

  hipMemsetAsync(cnt, 0, (size_t)N_ * R_ * 4, stream);
  hipMemsetAsync(relcnt, 0, R_ * 4, stream);

  k_entity<<<(B_ * H_ + 255) / 256, 256, 0, stream>>>(seq, e1m, e2m, e1h, e2h);
  dim3 geg(H_ / 256, N_);
  k_graph_embs<<<geg, 256, 0, stream>>>(seq, dep, nb, x0);
  k_count<<<(E_ + 255) / 256, 256, 0, stream>>>(ei, et, cnt, relcnt);
  k_scan<<<1, 32, 0, stream>>>(relcnt, offs, tiles, ntl, cursor);
  k_scatter<<<(E_ + 255) / 256, 256, 0, stream>>>(ei, et, cnt, offs, cursor, perm, norm);

  dim3 eg(NG_, MAXTILES);
  dim3 rg(NG_, MG_);
  // layer 1
  hipMemsetAsync(agg, 0, (size_t)N_ * H_ * 4, stream);
  k_edge_gemm<<<eg, 32, 0, stream>>>(x0, Wrel1, ei, perm, offs, tiles, ntl, norm, agg);
  k_root_gemm<<<rg, 32, 0, stream>>>(x0, Wr1, b1, agg, h1);
  // layer 2
  hipMemsetAsync(agg, 0, (size_t)N_ * H_ * 4, stream);
  k_edge_gemm<<<eg, 32, 0, stream>>>(h1, Wrel2, ei, perm, offs, tiles, ntl, norm, agg);
  k_root_gemm<<<rg, 32, 0, stream>>>(h1, Wr2, b2, agg, x0);   // h2 -> x0

  k_dep_pool<<<(B_ * H_ + 255) / 256, 256, 0, stream>>>(x0, n1, n2, e1d, e2d);
  k_final<<<(B_ * 3 * H_ + 255) / 256, 256, 0, stream>>>(pooled, e1h, e1d, e2h, e2d, rel_out);
  k_logits<<<(B_ * C_ + 255) / 256, 256, 0, stream>>>(rel_out, Wc, bc, logits);
}
